// SimpleViT_55465207660644
// MI455X (gfx1250) — matrix operand order repaired and checked by reading, not compiled, not run on hardware
//
#include <hip/hip_runtime.h>
#include <hip/hip_bf16.h>

typedef __attribute__((ext_vector_type(16))) _Float16 v16h;
typedef __attribute__((ext_vector_type(8)))  _Float16 v8h;
typedef __attribute__((ext_vector_type(8)))  float    v8f;

#define BB   32
#define TT   196
#define DD   768
#define NHH  12
#define HDD  64
#define LL   12
#define MLPD 3072
#define NCC  1000
#define TPAD 224   // 196 padded to multiple of 32

union H8  { v8h v;  _Float16 e[8]; };
union H16 { v16h v; v8h h[2]; };

// ---- WMMA fragment helpers (CDNA5 16x16x32 f16 layouts, ISA 7.12.2) ----
__device__ inline v16h frag_a(const _Float16* p, int lane) {
  const int koff = (lane & 16) ? 8 : 0;
  H16 u;
  u.h[0] = *(const v8h*)(p + koff);
  u.h[1] = *(const v8h*)(p + koff + 16);
  return u.v;
}
__device__ inline v16h frag_b(const _Float16* p, int lane) {
  const int koff = (lane & 16) ? 8 : 0;
  H16 u;
  u.h[0] = *(const v8h*)(p + koff);
  u.h[1] = *(const v8h*)(p + koff + 16);
  return u.v;
}
__device__ inline v8f wmma16(v16h a, v16h b, v8f c) {
  return __builtin_amdgcn_wmma_f32_16x16x32_f16(false, a, false, b, (short)0, c,
                                                false, false);
}
__device__ inline float gelu_f(float v) {
  return 0.5f * v * (1.0f + erff(v * 0.70710678118654752440f));
}
// Generic pointer -> LDS byte address (low 32 bits of shared-aperture addr)
__device__ inline unsigned lds_addr(const void* p) {
  return (unsigned)(unsigned long long)p;
}
// CDNA5 async copy: 16B global -> LDS, tracked by ASYNCcnt
__device__ inline void async_cp16(unsigned lds, const _Float16* g) {
  asm volatile("global_load_async_to_lds_b128 %0, %1, off"
               :: "v"(lds), "v"(g) : "memory");
}

// =======================================================================
// WMMA GEMM: C[M,N] = A_f16[M,K] @ Bt_f16[N][K] (+bias,+gelu,+resid)
// Tile 128x128x32, 256 thr = 8 waves, wave tile 64x32 (4x2 WMMA tiles).
// Double-buffered LDS staged with global_load_async_to_lds_b128.
// =======================================================================
template<int ACT, int RESID, int OUTF16>
__global__ void __launch_bounds__(256)
gemm_wmma_kernel(const _Float16* __restrict__ A,
                 const _Float16* __restrict__ Bt,
                 const float* __restrict__ bias,
                 const float* resid,
                 float* outF, _Float16* outH,
                 int M, int N, int K)
{
  __shared__ __align__(16) _Float16 As[2][128 * 32];
  __shared__ __align__(16) _Float16 Bs[2][128 * 32];

  const int tid   = threadIdx.x;
  const int lane  = tid & 31;
  const int wave  = tid >> 5;
  const int wm    = (wave & 1) * 64;
  const int wn    = (wave >> 1) * 32;
  const int m0    = blockIdx.y * 128;
  const int n0    = blockIdx.x * 128;
  const int ml    = lane & 15;
  const int rbase = (lane & 16) ? 8 : 0;
  const bool fullTile = (m0 + 128 <= M) && (n0 + 128 <= N);

  const v8f zf = {0.f,0.f,0.f,0.f,0.f,0.f,0.f,0.f};
  const v8h hz = {(_Float16)0,(_Float16)0,(_Float16)0,(_Float16)0,
                  (_Float16)0,(_Float16)0,(_Float16)0,(_Float16)0};
  v8f acc[4][2];
#pragma unroll
  for (int i = 0; i < 4; i++)
#pragma unroll
    for (int j = 0; j < 2; j++) acc[i][j] = zf;

  auto stage = [&](int t, int buf) {
    const int k0 = t * 32;
    if (fullTile) {
      // 4 async B128 issues per thread (2 A-chunks + 2 B-chunks)
#pragma unroll
      for (int u = 0; u < 2; u++) {
        const int chunk = tid + u * 256;       // 0..511 (16-B chunks)
        const int r = chunk >> 2;              // 0..127
        const int c = (chunk & 3) * 8;         // 0,8,16,24 halfs
        async_cp16(lds_addr(&As[buf][r * 32 + c]), A  + (long)(m0 + r) * K + k0 + c);
        async_cp16(lds_addr(&Bs[buf][r * 32 + c]), Bt + (long)(n0 + r) * K + k0 + c);
      }
    } else {
      const int r  = tid >> 1;
      const int c  = (tid & 1) * 16;
      const int gm = m0 + r;
      const int gn = n0 + r;
      v8h a0 = hz, a1 = hz, b0 = hz, b1 = hz;
      if (gm < M) {
        const _Float16* s = A + (long)gm * K + k0 + c;
        a0 = *(const v8h*)s; a1 = *(const v8h*)(s + 8);
      }
      if (gn < N) {
        const _Float16* s = Bt + (long)gn * K + k0 + c;
        b0 = *(const v8h*)s; b1 = *(const v8h*)(s + 8);
      }
      *(v8h*)(&As[buf][r * 32 + c])     = a0;
      *(v8h*)(&As[buf][r * 32 + c + 8]) = a1;
      *(v8h*)(&Bs[buf][r * 32 + c])     = b0;
      *(v8h*)(&Bs[buf][r * 32 + c + 8]) = b1;
    }
  };

  const int nk = K >> 5;
  stage(0, 0);
  for (int t = 0; t < nk; t++) {
    const int cur = t & 1;
    const bool hasNext = (t + 1 < nk);
    if (hasNext) stage(t + 1, cur ^ 1);
    if (hasNext) asm volatile("s_wait_asynccnt 0x4" ::: "memory");
    else         asm volatile("s_wait_asynccnt 0x0" ::: "memory");
    __syncthreads();
    v16h af[4], bf[2];
#pragma unroll
    for (int i = 0; i < 4; i++) af[i] = frag_a(&As[cur][(wm + i * 16 + ml) * 32], lane);
#pragma unroll
    for (int j = 0; j < 2; j++) bf[j] = frag_b(&Bs[cur][(wn + j * 16 + ml) * 32], lane);
#pragma unroll
    for (int i = 0; i < 4; i++)
#pragma unroll
      for (int j = 0; j < 2; j++)
        acc[i][j] = wmma16(af[i], bf[j], acc[i][j]);
    __syncthreads();
  }

  // epilogue (C layout: lane -> N = lane&15; VGPR r -> M = r + rbase)
#pragma unroll
  for (int i = 0; i < 4; i++) {
#pragma unroll
    for (int j = 0; j < 2; j++) {
      const int gn = n0 + wn + j * 16 + ml;
      if (gn >= N) continue;
      const float bv = bias ? bias[gn] : 0.0f;
#pragma unroll
      for (int r = 0; r < 8; r++) {
        const int gm = m0 + wm + i * 16 + rbase + r;
        if (gm >= M) continue;
        float vv = acc[i][j][r] + bv;
        if (ACT == 1) vv = gelu_f(vv);
        if (RESID) vv += resid[(long)gm * N + gn];
        if (OUTF16) outH[(long)gm * N + gn] = (_Float16)vv;
        else        outF[(long)gm * N + gn] = vv;
      }
    }
  }
}

// =======================================================================
// Attention: one workgroup per (batch, head). S = QK^T*scale, softmax,
// O = P V.  T=196 padded to 224. Dynamic LDS ~154KB (vsT overlays qs).
// Q/K tiles staged with async LDS copies.
// =======================================================================
__global__ void __launch_bounds__(256)
attention_kernel(const _Float16* __restrict__ q,
                 const _Float16* __restrict__ k,
                 const _Float16* __restrict__ v,
                 _Float16* __restrict__ o)
{
  extern __shared__ __align__(16) _Float16 smem[];
  _Float16* qs  = smem;                   // [224][64]
  _Float16* ks  = smem + TPAD * HDD;      // [224][64]
  _Float16* vsT = smem;                   // [64][224]  (overlays qs, phase 2)
  _Float16* ps  = smem + 2 * TPAD * HDD;  // [224][224] scores -> probs

  const int tid   = threadIdx.x;
  const int lane  = tid & 31;
  const int wave  = tid >> 5;
  const int ml    = lane & 15;
  const int rbase = (lane & 16) ? 8 : 0;
  const int bh    = blockIdx.x;
  const int b     = bh / NHH;
  const int hh    = bh % NHH;
  const long base = ((long)b * TT) * DD + hh * HDD;

  const v8f zf = {0.f,0.f,0.f,0.f,0.f,0.f,0.f,0.f};
  const v8h hz = {(_Float16)0,(_Float16)0,(_Float16)0,(_Float16)0,
                  (_Float16)0,(_Float16)0,(_Float16)0,(_Float16)0};

  // ---- async-stage Q and K rows [0,196); zero-fill pad rows [196,224) ----
  for (int chunk = tid; chunk < TT * 8; chunk += 256) {   // 16-B chunks
    const int row = chunk >> 3;
    const int c   = (chunk & 7) * 8;
    async_cp16(lds_addr(qs + row * 64 + c), q + base + (long)row * DD + c);
    async_cp16(lds_addr(ks + row * 64 + c), k + base + (long)row * DD + c);
  }
  for (int it = tid; it < (TPAD - TT) * 4; it += 256) {   // 28 rows x 4 chunks
    const int row = TT + (it >> 2);
    const int c   = (it & 3) * 16;
    *(v8h*)(qs + row * 64 + c)     = hz;
    *(v8h*)(qs + row * 64 + c + 8) = hz;
    *(v8h*)(ks + row * 64 + c)     = hz;
    *(v8h*)(ks + row * 64 + c + 8) = hz;
  }
  asm volatile("s_wait_asynccnt 0x0" ::: "memory");
  __syncthreads();

  // ---- phase 1: S = Q K^T * scale  (14x14 tiles of 16x16, K=64) ----
  const float scale = 0.125f;   // HD^-0.5
  for (int t = wave; t < 196; t += 8) {        // uniform per wave: EXEC full
    const int ti = t / 14, tj = t % 14;
    v8f acc = zf;
#pragma unroll
    for (int kk = 0; kk < HDD; kk += 32) {
      v16h af = frag_a(qs + (ti * 16 + ml) * 64 + kk, lane);
      v16h bf = frag_b(ks + (tj * 16 + ml) * 64 + kk, lane);
      acc = wmma16(af, bf, acc);
    }
#pragma unroll
    for (int r = 0; r < 8; r++)
      ps[(ti * 16 + rbase + r) * TPAD + tj * 16 + ml] = (_Float16)(acc[r] * scale);
  }
  __syncthreads();

  // ---- load V transposed into vsT (overlays dead qs) + softmax on ps ----
  for (int it = tid; it < TPAD * 8; it += 256) {
    const int s  = it >> 3;
    const int e0 = (it & 7) * 8;
    H8 val; val.v = hz;
    if (s < TT) val.v = *(const v8h*)(v + base + (long)s * DD + e0);
#pragma unroll
    for (int j = 0; j < 8; j++) vsT[(e0 + j) * TPAD + s] = val.e[j];
  }
  if (tid < TT) {
    _Float16* row = ps + tid * TPAD;
    float mx = -1e30f;
    for (int j = 0; j < TT; j++) mx = fmaxf(mx, (float)row[j]);
    float s = 0.f;
    for (int j = 0; j < TT; j++) {
      const float e = __expf((float)row[j] - mx);
      s += e;
      row[j] = (_Float16)e;
    }
    const float inv = 1.0f / s;
    for (int j = 0; j < TT; j++) row[j] = (_Float16)((float)row[j] * inv);
    for (int j = TT; j < TPAD; j++) row[j] = (_Float16)0;
  }
  __syncthreads();

  // ---- phase 2: O = P V  (14 M-tiles x 4 N-tiles, K=224 in 7 steps) ----
  for (int t = wave; t < 56; t += 8) {
    const int ti = t / 4, tj = t % 4;
    v8f acc = zf;
#pragma unroll
    for (int kk = 0; kk < TPAD; kk += 32) {
      v16h af = frag_a(ps  + (ti * 16 + ml) * TPAD + kk, lane);
      v16h bf = frag_b(vsT + (tj * 16 + ml) * TPAD + kk, lane);
      acc = wmma16(af, bf, acc);
    }
#pragma unroll
    for (int r = 0; r < 8; r++) {
      const int gm = ti * 16 + rbase + r;
      if (gm < TT)
        o[base + (long)gm * DD + tj * 16 + ml] = (_Float16)acc[r];
    }
  }
}

// ============================ small kernels ============================
// Wt[n][k] (f16) = W[k*strideK + (n>>6)*strideH + (n&63)] (fp32)
__global__ void convert_weight_kernel(const float* __restrict__ W,
                                      _Float16* __restrict__ Wt,
                                      int N, int K, int strideK, int strideH)
{
  const long idx = (long)blockIdx.x * 256 + threadIdx.x;
  if (idx >= (long)N * K) return;
  const int kk = (int)(idx / N);
  const int n  = (int)(idx % N);
  Wt[(long)n * K + kk] =
      (_Float16)W[(long)kk * strideK + (long)(n >> 6) * strideH + (n & 63)];
}

__global__ void patchify_kernel(const float* __restrict__ x,
                                _Float16* __restrict__ patches)
{
  const int idx = blockIdx.x * 256 + threadIdx.x;
  if (idx >= BB * TT * DD) return;        // PATCH_DIM == 768 == DD
  const int col = idx % DD;
  const int row = idx / DD;
  const int b   = row / TT;
  const int t   = row % TT;
  const int gh  = t / 14, gw = t % 14;
  const int p1  = col / 48;
  const int rem = col % 48;
  const int p2  = rem / 3;
  const int c   = rem % 3;
  const long src = (((long)b * 3 + c) * 224 + (gh * 16 + p1)) * 224 + (gw * 16 + p2);
  patches[idx] = (_Float16)x[src];
}

__global__ void posemb_kernel(float* h)
{
  const int idx = blockIdx.x * 256 + threadIdx.x;
  if (idx >= BB * TT * DD) return;
  const int d = idx % DD;
  const int t = (idx / DD) % TT;
  const float y = (float)(t / 14);
  const float x = (float)(t % 14);
  const int quad = d / 192;
  const int j    = d % 192;
  const float omega = __powf(10000.0f, -(float)j / 191.0f);
  float pe;
  if      (quad == 0) pe = __sinf(x * omega);
  else if (quad == 1) pe = __cosf(x * omega);
  else if (quad == 2) pe = __sinf(y * omega);
  else                pe = __cosf(y * omega);
  h[idx] += pe;
}

__global__ void __launch_bounds__(256)
layernorm_kernel(const float* __restrict__ h, const float* __restrict__ g,
                 const float* __restrict__ bt, _Float16* __restrict__ xn, int rows)
{
  const int lane = threadIdx.x & 31;
  const int wave = threadIdx.x >> 5;
  const int row  = blockIdx.x * 8 + wave;
  if (row >= rows) return;
  const float* p = h + (long)row * DD;
  float s = 0.f;
  for (int i = lane; i < DD; i += 32) s += p[i];
#pragma unroll
  for (int off = 16; off > 0; off >>= 1) s += __shfl_xor(s, off, 32);
  const float mu = s / (float)DD;
  float var = 0.f;
  for (int i = lane; i < DD; i += 32) { const float d = p[i] - mu; var += d * d; }
#pragma unroll
  for (int off = 16; off > 0; off >>= 1) var += __shfl_xor(var, off, 32);
  const float rstd = rsqrtf(var / (float)DD + 1e-5f);
  _Float16* op = xn + (long)row * DD;
  for (int i = lane; i < DD; i += 32)
    op[i] = (_Float16)((p[i] - mu) * rstd * g[i] + bt[i]);
}

__global__ void mean_kernel(const float* __restrict__ h, _Float16* __restrict__ hm)
{
  const int idx = blockIdx.x * 256 + threadIdx.x;
  if (idx >= BB * DD) return;
  const int b = idx / DD, d = idx % DD;
  float s = 0.f;
  for (int t = 0; t < TT; t++) s += h[((long)b * TT + t) * DD + d];
  hm[idx] = (_Float16)(s / (float)TT);
}

// ================================ host =================================
static inline void convert_w(const float* W, _Float16* Wt, int N, int K,
                             int sK, int sH, hipStream_t stream)
{
  const long tot = (long)N * K;
  convert_weight_kernel<<<(unsigned)((tot + 255) / 256), 256, 0, stream>>>(
      W, Wt, N, K, sK, sH);
}

extern "C" void kernel_launch(void* const* d_in, const int* in_sizes, int n_in,
                              void* d_out, int out_size, void* d_ws, size_t ws_size,
                              hipStream_t stream)
{
  (void)in_sizes; (void)n_in; (void)out_size; (void)ws_size;
  const float* x       = (const float*)d_in[0];
  const float* embed_W = (const float*)d_in[1];
  const float* embed_b = (const float*)d_in[2];
  const float* Wq      = (const float*)d_in[3];
  const float* Wk      = (const float*)d_in[4];
  const float* Wv      = (const float*)d_in[5];
  const float* Wo      = (const float*)d_in[6];
  const float* bo      = (const float*)d_in[7];
  const float* ln1_g   = (const float*)d_in[8];
  const float* ln1_b   = (const float*)d_in[9];
  const float* ln2_g   = (const float*)d_in[10];
  const float* ln2_b   = (const float*)d_in[11];
  const float* W1      = (const float*)d_in[12];
  const float* b1      = (const float*)d_in[13];
  const float* W2      = (const float*)d_in[14];
  const float* b2      = (const float*)d_in[15];
  const float* head_W  = (const float*)d_in[16];

  const int M = BB * TT;                       // 6272 (multiple of 128)
  char* ws = (char*)d_ws;
  _Float16* patches = (_Float16*)ws; ws += (long)M * DD * 2;
  float*    h       = (float*)ws;    ws += (long)M * DD * 4;
  _Float16* xn      = (_Float16*)ws; ws += (long)M * DD * 2;
  _Float16* qb      = (_Float16*)ws; ws += (long)M * DD * 2;
  _Float16* kb      = (_Float16*)ws; ws += (long)M * DD * 2;
  _Float16* vb      = (_Float16*)ws; ws += (long)M * DD * 2;
  _Float16* attn    = (_Float16*)ws; ws += (long)M * DD * 2;
  _Float16* mlp     = (_Float16*)ws; ws += (long)M * MLPD * 2;
  _Float16* hm      = (_Float16*)ws; ws += (long)BB * DD * 2;
  // f16 transposed weights [N][K]
  _Float16* embedT = (_Float16*)ws;  ws += (long)DD * DD * 2;
  _Float16* headT  = (_Float16*)ws;  ws += (long)NCC * DD * 2;
  _Float16* wqT[LL], *wkT[LL], *wvT[LL], *woT[LL], *w1T[LL], *w2T[LL];
  for (int l = 0; l < LL; l++) {
    wqT[l] = (_Float16*)ws; ws += (long)DD * DD * 2;
    wkT[l] = (_Float16*)ws; ws += (long)DD * DD * 2;
    wvT[l] = (_Float16*)ws; ws += (long)DD * DD * 2;
    woT[l] = (_Float16*)ws; ws += (long)DD * DD * 2;
    w1T[l] = (_Float16*)ws; ws += (long)MLPD * DD * 2;
    w2T[l] = (_Float16*)ws; ws += (long)DD * MLPD * 2;
  }

  // ---- one-pass weight convert/transpose to f16 [N][K] ----
  convert_w(embed_W, embedT, DD, DD, DD, 64, stream);
  convert_w(head_W,  headT, NCC, DD, NCC, 64, stream);
  for (int l = 0; l < LL; l++) {
    convert_w(Wq + (long)l * DD * DD,   wqT[l], DD, DD, HDD, DD * HDD, stream);
    convert_w(Wk + (long)l * DD * DD,   wkT[l], DD, DD, HDD, DD * HDD, stream);
    convert_w(Wv + (long)l * DD * DD,   wvT[l], DD, DD, HDD, DD * HDD, stream);
    convert_w(Wo + (long)l * DD * DD,   woT[l], DD, DD, DD, 64, stream);
    convert_w(W1 + (long)l * DD * MLPD, w1T[l], MLPD, DD, MLPD, 64, stream);
    convert_w(W2 + (long)l * MLPD * DD, w2T[l], DD, MLPD, DD, 64, stream);
  }

  const dim3 blk(256);
  const dim3 g768((DD + 127) / 128,  (M + 127) / 128);
  const dim3 gmlp((MLPD + 127) / 128, (M + 127) / 128);
  const dim3 ghead((NCC + 127) / 128, 1);
  const size_t SMEM = (size_t)(2 * TPAD * HDD + TPAD * TPAD) * sizeof(_Float16);

  patchify_kernel<<<(M * DD + 255) / 256, blk, 0, stream>>>(x, patches);
  gemm_wmma_kernel<0,0,0><<<g768, blk, 0, stream>>>(
      patches, embedT, embed_b, nullptr, h, nullptr, M, DD, DD);
  posemb_kernel<<<(M * DD + 255) / 256, blk, 0, stream>>>(h);

  for (int l = 0; l < LL; l++) {
    layernorm_kernel<<<(M + 7) / 8, blk, 0, stream>>>(h, ln1_g + l * DD,
                                                      ln1_b + l * DD, xn, M);
    gemm_wmma_kernel<0,0,1><<<g768, blk, 0, stream>>>(
        xn, wqT[l], nullptr, nullptr, nullptr, qb, M, DD, DD);
    gemm_wmma_kernel<0,0,1><<<g768, blk, 0, stream>>>(
        xn, wkT[l], nullptr, nullptr, nullptr, kb, M, DD, DD);
    gemm_wmma_kernel<0,0,1><<<g768, blk, 0, stream>>>(
        xn, wvT[l], nullptr, nullptr, nullptr, vb, M, DD, DD);
    attention_kernel<<<BB * NHH, blk, SMEM, stream>>>(qb, kb, vb, attn);
    gemm_wmma_kernel<0,1,0><<<g768, blk, 0, stream>>>(
        attn, woT[l], bo + l * DD, h, h, nullptr, M, DD, DD);
    layernorm_kernel<<<(M + 7) / 8, blk, 0, stream>>>(h, ln2_g + l * DD,
                                                      ln2_b + l * DD, xn, M);
    gemm_wmma_kernel<1,0,1><<<gmlp, blk, 0, stream>>>(
        xn, w1T[l], b1 + l * MLPD, nullptr, nullptr, mlp, M, MLPD, DD);
    gemm_wmma_kernel<0,1,0><<<g768, blk, 0, stream>>>(
        mlp, w2T[l], b2 + l * DD, h, h, nullptr, M, DD, MLPD);
  }

  mean_kernel<<<(BB * DD + 255) / 256, blk, 0, stream>>>(h, hm);
  gemm_wmma_kernel<0,0,0><<<ghead, blk, 0, stream>>>(
      hm, headT, nullptr, nullptr, (float*)d_out, nullptr, BB, NCC, DD);
}
